// Block_9165460210159
// MI455X (gfx1250) — matrix operand order repaired and checked by reading, not compiled, not run on hardware
//
#include <hip/hip_runtime.h>
#include <hip/hip_bf16.h>
#include <math.h>

// ---------------- constants from the reference ----------------
#define DIM      256
#define HID      1024
#define D_STATE  128
#define HEADDIM  64
#define D_INNER  512            // 2*DIM
#define NHEADS   8              // D_INNER/HEADDIM
#define CONV_DIM 768            // D_INNER + 2*D_STATE
#define D_IN_PROJ 1288          // 2*D_INNER + 2*D_STATE + NHEADS
#define D_IP_PAD 1344           // padded to multiple of 64 for GEMM tiling
#define KCONV    4
#define BB       8
#define NN       4096
#define KNN_K    16
#define TOK      (BB*NN)        // 32768 tokens
#define EPSV     1e-5f

typedef __bf16 bf16;
typedef __attribute__((ext_vector_type(16))) __bf16 v16bf;
typedef __attribute__((ext_vector_type(8)))  __bf16 v8bf;
typedef __attribute__((ext_vector_type(8)))  float  v8f;

#define LDSPAD 40   // bf16 elements per staged row (32 data + 8 pad = 80B, kills bank conflicts)

// ---------------- WMMA GEMM: C[M x Nld] = A[M x K] * Bt[Nld x K]^T ----------------
// Block = 256 threads = 8 waves. Block tile 256(M) x 64(N); wave tile 32 x 64
// (2x4 WMMA f32 tiles). K-step 32. A/B tiles double-buffered in LDS, filled with
// GLOBAL_LOAD_ASYNC_TO_LDS_B128 (ASYNCcnt) so the next stage overlaps WMMA.
__global__ __launch_bounds__(256) void k_gemm(const bf16* __restrict__ A,
                                              const bf16* __restrict__ Bt,
                                              float* __restrict__ C,
                                              int M, int Nld, int K)
{
    __shared__ bf16 sA[2][256][LDSPAD];   // 40 KB
    __shared__ bf16 sB[2][64][LDSPAD];    // 10 KB

    const int tid  = threadIdx.x;
    const int lane = tid & 31;
    const int w    = tid >> 5;          // wave 0..7
    const int hl   = lane >> 4;         // half of wave: 0/1
    const int r    = lane & 15;         // row/col within 16
    const int mblk = blockIdx.y * 256;
    const int nblk = blockIdx.x * 64;
    const int m0w  = mblk + w * 32;     // this wave's M origin

    const int rowT = tid >> 2;          // 0..63: staging row group
    const int seg  = tid & 3;           // 16B segment within a 64B row

    // issue one double-buffer stage: A tile 256x32 (4 xfers/thread), B tile 64x32 (1 xfer/thread)
    auto issue = [&](int buf, int k0) {
        #pragma unroll
        for (int i = 0; i < 4; ++i) {
            int row = rowT + i * 64;
            const bf16* gp = A + (size_t)(mblk + row) * K + k0 + seg * 8;
            unsigned lo = (unsigned)(size_t)&sA[buf][row][seg * 8];
            asm volatile("global_load_async_to_lds_b128 %0, %1, off"
                         :: "v"(lo), "v"(gp) : "memory");
        }
        {
            const bf16* gp = Bt + (size_t)(nblk + rowT) * K + k0 + seg * 8;
            unsigned lo = (unsigned)(size_t)&sB[buf][rowT][seg * 8];
            asm volatile("global_load_async_to_lds_b128 %0, %1, off"
                         :: "v"(lo), "v"(gp) : "memory");
        }
    };

    v8f acc[2][4] = {};

    const int ksteps = K >> 5;
    issue(0, 0);
    for (int kt = 0; kt < ksteps; ++kt) {
        int buf = kt & 1;
        if (kt + 1 < ksteps) {
            issue(buf ^ 1, (kt + 1) << 5);
            asm volatile("s_wait_asynccnt 0x5" ::: "memory"); // only the 5 just-issued remain
        } else {
            asm volatile("s_wait_asynccnt 0x0" ::: "memory");
        }
        __syncthreads();

        v16bf a[2], b[4];
        // A fragment (16x32, ISA layout): lane<16 -> K {0..7,16..23}, lane>=16 -> K {8..15,24..31}
        #pragma unroll
        for (int i = 0; i < 2; ++i) {
            const bf16* ap = &sA[buf][w * 32 + i * 16 + r][0];
            *(v8bf*)&a[i]       = *(const v8bf*)(ap + hl * 8);
            *((v8bf*)&a[i] + 1) = *(const v8bf*)(ap + 16 + hl * 8);
        }
        // B fragment (32x16): lane<16 holds K {0..7,16..23} of its column, lane>=16 K {8..15,24..31}
        #pragma unroll
        for (int j = 0; j < 4; ++j) {
            const bf16* bp = &sB[buf][j * 16 + r][0];
            *(v8bf*)&b[j]       = *(const v8bf*)(bp + hl * 8);
            *((v8bf*)&b[j] + 1) = *(const v8bf*)(bp + 16 + hl * 8);
        }
        #pragma unroll
        for (int i = 0; i < 2; ++i)
            #pragma unroll
            for (int j = 0; j < 4; ++j)
                acc[i][j] = __builtin_amdgcn_wmma_f32_16x16x32_bf16(
                    false, a[i], false, b[j], (short)0, acc[i][j], false, false);
        __syncthreads();
    }

    // C/D layout: VGPR g holds row (g + hl*8), col r of the 16x16 tile.
    // Lanes 0..15 of a half-wave cover 16 consecutive columns -> coalesced 64B bursts.
    #pragma unroll
    for (int i = 0; i < 2; ++i)
        #pragma unroll
        for (int j = 0; j < 4; ++j) {
            float* cp = C + (size_t)(m0w + i * 16 + hl * 8) * Nld + (nblk + j * 16 + r);
            #pragma unroll
            for (int g = 0; g < 8; ++g)
                cp[(size_t)g * Nld] = acc[i][j][g];
        }
    (void)M;
}

// ---------------- weight transpose + fp32->bf16 (with zero row padding) ----------------
// W[K x Nn] fp32 -> Wt[Nld x K] bf16 (rows n >= Nn zero-filled)
__global__ __launch_bounds__(256) void k_wcvt(const float* __restrict__ W,
                                              bf16* __restrict__ Wt,
                                              int K, int Nn, int Nld)
{
    int i = blockIdx.x * 256 + threadIdx.x;
    int total = Nld * K;
    if (i >= total) return;
    int n = i / K, k = i - n * K;
    float v = (n < Nn) ? W[(size_t)k * Nn + n] : 0.f;
    Wt[i] = (bf16)v;
}

// ---------------- fp32 -> bf16 ----------------
__global__ __launch_bounds__(256) void k_cvt(const float* __restrict__ X,
                                             bf16* __restrict__ Y, int total)
{
    int i = blockIdx.x * 256 + threadIdx.x;
    if (i < total) Y[i] = (bf16)X[i];
}

// ---------------- bias + gelu(tanh) -> bf16 ----------------
__global__ __launch_bounds__(256) void k_bias_gelu(const float* __restrict__ H,
                                                   const float* __restrict__ bias,
                                                   bf16* __restrict__ O,
                                                   int C, int total)
{
    int i = blockIdx.x * 256 + threadIdx.x;
    if (i >= total) return;
    int c = i % C;
    float v = H[i] + bias[c];
    float t3 = v * v * v;
    float g = 0.5f * v * (1.f + tanhf(0.7978845608028654f * (v + 0.044715f * t3)));
    O[i] = (bf16)g;
}

// ---------------- batchnorm + residual add ----------------
__global__ __launch_bounds__(256) void k_bn_add(float* __restrict__ X,
                                                const float* __restrict__ T,
                                                const float* __restrict__ g,
                                                const float* __restrict__ b,
                                                const float* __restrict__ m,
                                                const float* __restrict__ v,
                                                int C, int total)
{
    int i = blockIdx.x * 256 + threadIdx.x;
    if (i >= total) return;
    int c = i % C;
    X[i] += (T[i] - m[c]) * rsqrtf(v[c] + EPSV) * g[c] + b[c];
}

// ---------------- kNN max-pool (gather) + BN + residual ----------------
__global__ __launch_bounds__(256) void k_knn_bn_add(float* __restrict__ X,
                                                    const float* __restrict__ XW,
                                                    const int* __restrict__ KNN,
                                                    const float* __restrict__ g,
                                                    const float* __restrict__ b,
                                                    const float* __restrict__ m,
                                                    const float* __restrict__ v,
                                                    int total)
{
    int i = blockIdx.x * 256 + threadIdx.x;
    if (i >= total) return;
    int d = i & (DIM - 1);
    int token = i >> 8;
    int bb = token >> 12;
    int n  = token & (NN - 1);
    float base = XW[(size_t)token * DIM + d];
    const int* kp = KNN + ((size_t)bb * NN + n) * KNN_K;
    float mx = -1e30f;
    #pragma unroll
    for (int k = 0; k < KNN_K; ++k) {
        int idx = kp[k];
        float vv = XW[((size_t)(bb << 12) + idx) * DIM + d];
        mx = fmaxf(mx, vv);
    }
    float t = mx - base;
    X[i] += (t - m[d]) * rsqrtf(v[d] + EPSV) * g[d] + b[d];
}

// ---------------- RMS norm (C=256) -> bf16, one wave per token ----------------
__global__ __launch_bounds__(256) void k_rms_cvt(const float* __restrict__ X,
                                                 const float* __restrict__ w,
                                                 bf16* __restrict__ O)
{
    int wave = threadIdx.x >> 5, lane = threadIdx.x & 31;
    int token = blockIdx.x * 8 + wave;
    const float* xr = X + (size_t)token * DIM;
    float vals[8]; float ss = 0.f;
    #pragma unroll
    for (int j = 0; j < 8; ++j) {
        int c = lane + j * 32;
        float v = xr[c];
        vals[j] = v; ss += v * v;
    }
    for (int o = 16; o > 0; o >>= 1) ss += __shfl_xor(ss, o, 32);
    float sc = rsqrtf(ss * (1.f / DIM) + EPSV);
    #pragma unroll
    for (int j = 0; j < 8; ++j) {
        int c = lane + j * 32;
        O[(size_t)token * DIM + c] = (bf16)(vals[j] * sc * w[c]);
    }
}

// ---------------- depthwise causal conv(K=4) + SiLU -> bf16 ----------------
__global__ __launch_bounds__(256) void k_conv_silu(const float* __restrict__ ZX,
                                                   const float* __restrict__ cw,
                                                   const float* __restrict__ cb,
                                                   bf16* __restrict__ XBC,
                                                   int total)
{
    int i = blockIdx.x * 256 + threadIdx.x;
    if (i >= total) return;
    int c = i % CONV_DIM;
    int token = i / CONV_DIM;
    int bb = token >> 12;
    int n  = token & (NN - 1);
    float acc = 0.f;
    #pragma unroll
    for (int j = 0; j < KCONV; ++j) {
        int nn = n + j - (KCONV - 1);
        if (nn >= 0)
            acc += cw[c * KCONV + j] *
                   ZX[((size_t)(bb << 12) + nn) * D_IP_PAD + D_INNER + c];
    }
    acc += cb[c];
    float s = acc / (1.f + __expf(-acc));
    XBC[i] = (bf16)s;
}

// ---------------- dt = softplus(dt + bias) ----------------
__global__ __launch_bounds__(256) void k_dt(const float* __restrict__ ZX,
                                            const float* __restrict__ dtb,
                                            float* __restrict__ DT, int total)
{
    int i = blockIdx.x * 256 + threadIdx.x;
    if (i >= total) return;
    int h = i & (NHEADS - 1);
    int token = i >> 3;
    float v = ZX[(size_t)token * D_IP_PAD + (D_INNER + CONV_DIM) + h] + dtb[h];
    DT[i] = (v > 20.f) ? v : log1pf(__expf(v));
}

// ---------------- SSM sequential scan, one block per (batch, head) ----------------
__global__ __launch_bounds__(256) void k_scan(const bf16* __restrict__ XBC,
                                              const float* __restrict__ DT,
                                              const float* __restrict__ A_log,
                                              const float* __restrict__ Dp,
                                              float* __restrict__ Y)
{
    __shared__ float sx[HEADDIM];
    __shared__ float sB[D_STATE];
    __shared__ float sC[D_STATE];
    int bh = blockIdx.x;
    int bb = bh >> 3, h = bh & 7;
    int tid = threadIdx.x;
    float Ah = -__expf(A_log[h]);
    float Dh = Dp[h];
    int p = tid >> 2, q = tid & 3, nb = q << 5;
    float hreg[32];
    #pragma unroll
    for (int j = 0; j < 32; ++j) hreg[j] = 0.f;

    for (int t = 0; t < NN; ++t) {
        size_t base = ((size_t)(bb << 12) + t) * CONV_DIM;
        if (tid < HEADDIM) sx[tid] = (float)XBC[base + h * HEADDIM + tid];
        if (tid < D_STATE) sB[tid] = (float)XBC[base + D_INNER + tid];
        else               sC[tid - D_STATE] = (float)XBC[base + D_INNER + D_STATE + (tid - D_STATE)];
        __syncthreads();

        float dtt  = DT[((size_t)(bb << 12) + t) * NHEADS + h];
        float dA   = __expf(dtt * Ah);
        float coef = dtt * sx[p];
        float acc  = 0.f;
        #pragma unroll
        for (int j = 0; j < 32; ++j) {
            int n = nb + j;
            hreg[j] = hreg[j] * dA + coef * sB[n];
            acc += hreg[j] * sC[n];
        }
        acc += __shfl_xor(acc, 1, 4);
        acc += __shfl_xor(acc, 2, 4);
        if (q == 0)
            Y[((size_t)(bb << 12) + t) * D_INNER + h * HEADDIM + p] = acc + Dh * sx[p];
        __syncthreads();
    }
}

// ---------------- y = rms(y * silu(z)) * norm -> bf16 (C=512), one wave per token ----------------
__global__ __launch_bounds__(256) void k_gate_rms(const float* __restrict__ Y,
                                                  const float* __restrict__ ZX,
                                                  const float* __restrict__ nrm,
                                                  bf16* __restrict__ O)
{
    int wave = threadIdx.x >> 5, lane = threadIdx.x & 31;
    int token = blockIdx.x * 8 + wave;
    const float* yr = Y  + (size_t)token * D_INNER;
    const float* zr = ZX + (size_t)token * D_IP_PAD;   // z occupies cols [0,512)
    float vals[16]; float ss = 0.f;
    #pragma unroll
    for (int j = 0; j < 16; ++j) {
        int c = lane + j * 32;
        float z = zr[c];
        float sz = z / (1.f + __expf(-z));
        float v = yr[c] * sz;
        vals[j] = v; ss += v * v;
    }
    for (int o = 16; o > 0; o >>= 1) ss += __shfl_xor(ss, o, 32);
    float sc = rsqrtf(ss * (1.f / D_INNER) + EPSV);
    #pragma unroll
    for (int j = 0; j < 16; ++j) {
        int c = lane + j * 32;
        O[(size_t)token * D_INNER + c] = (bf16)(vals[j] * sc * nrm[c]);
    }
}

// ---------------- residual add ----------------
__global__ __launch_bounds__(256) void k_add(float* __restrict__ X,
                                             const float* __restrict__ T, int total)
{
    int i = blockIdx.x * 256 + threadIdx.x;
    if (i < total) X[i] += T[i];
}

// ================= host side =================
#define EWL(kern, total, ...) kern<<<((total) + 255) / 256, 256, 0, stream>>>(__VA_ARGS__)

extern "C" void kernel_launch(void* const* d_in, const int* in_sizes, int n_in,
                              void* d_out, int out_size, void* d_ws, size_t ws_size,
                              hipStream_t stream)
{
    const float* x_in      = (const float*)d_in[0];
    const int*   knn       = (const int*)  d_in[1];
    const float* mlp0_w1   = (const float*)d_in[2];
    const float* mlp0_b1   = (const float*)d_in[3];
    const float* mlp0_w2   = (const float*)d_in[4];
    const float* mlp0_g    = (const float*)d_in[5];
    const float* mlp0_b    = (const float*)d_in[6];
    const float* mlp0_m    = (const float*)d_in[7];
    const float* mlp0_v    = (const float*)d_in[8];
    const float* lfp_w     = (const float*)d_in[9];
    const float* lfp_g     = (const float*)d_in[10];
    const float* lfp_b     = (const float*)d_in[11];
    const float* lfp_m     = (const float*)d_in[12];
    const float* lfp_v     = (const float*)d_in[13];
    const float* mlps_w1   = (const float*)d_in[14];
    const float* mlps_b1   = (const float*)d_in[15];
    const float* mlps_w2   = (const float*)d_in[16];
    const float* mlps_g    = (const float*)d_in[17];
    const float* mlps_b    = (const float*)d_in[18];
    const float* mlps_m    = (const float*)d_in[19];
    const float* mlps_v    = (const float*)d_in[20];
    const float* mb_norm   = (const float*)d_in[21];
    const float* in_proj_w = (const float*)d_in[22];
    const float* conv_w    = (const float*)d_in[23];
    const float* conv_b    = (const float*)d_in[24];
    const float* dt_bias   = (const float*)d_in[25];
    const float* A_log     = (const float*)d_in[26];
    const float* Dp        = (const float*)d_in[27];
    const float* ssm_norm  = (const float*)d_in[28];
    const float* out_proj_w= (const float*)d_in[29];
    (void)in_sizes; (void)n_in; (void)out_size; (void)ws_size;

    float* X = (float*)d_out;   // running activation [TOK x 256]

    char* ws = (char*)d_ws;
    size_t off = 0;
    auto carve = [&](size_t bytes) -> void* {
        void* p = ws + off;
        off = (off + bytes + 255) & ~(size_t)255;
        return p;
    };
    bf16* wt_mlp0_w1 = (bf16*)carve((size_t)HID * DIM * 2);
    bf16* wt_mlp0_w2 = (bf16*)carve((size_t)DIM * HID * 2);
    bf16* wt_lfp     = (bf16*)carve((size_t)4 * DIM * DIM * 2);
    bf16* wt_mlps_w1 = (bf16*)carve((size_t)2 * HID * DIM * 2);
    bf16* wt_mlps_w2 = (bf16*)carve((size_t)2 * DIM * HID * 2);
    bf16* wt_ip      = (bf16*)carve((size_t)D_IP_PAD * DIM * 2);
    bf16* wt_op      = (bf16*)carve((size_t)DIM * D_INNER * 2);
    bf16* ab0        = (bf16*)carve((size_t)TOK * HID * 2);
    float* cf0       = (float*)carve((size_t)TOK * D_IP_PAD * 4);
    bf16* xbc        = (bf16*)carve((size_t)TOK * CONV_DIM * 2);
    float* dtv       = (float*)carve((size_t)TOK * NHEADS * 4);
    float* ybuf      = (float*)carve((size_t)TOK * D_INNER * 4);

    hipMemcpyAsync(X, x_in, (size_t)TOK * DIM * sizeof(float),
                   hipMemcpyDeviceToDevice, stream);

    EWL(k_wcvt, HID * DIM, mlp0_w1, wt_mlp0_w1, DIM, HID, HID);
    EWL(k_wcvt, DIM * HID, mlp0_w2, wt_mlp0_w2, HID, DIM, DIM);
    for (int i = 0; i < 4; ++i)
        EWL(k_wcvt, DIM * DIM, lfp_w + (size_t)i * DIM * DIM,
            wt_lfp + (size_t)i * DIM * DIM, DIM, DIM, DIM);
    for (int j = 0; j < 2; ++j) {
        EWL(k_wcvt, HID * DIM, mlps_w1 + (size_t)j * DIM * HID,
            wt_mlps_w1 + (size_t)j * HID * DIM, DIM, HID, HID);
        EWL(k_wcvt, DIM * HID, mlps_w2 + (size_t)j * HID * DIM,
            wt_mlps_w2 + (size_t)j * DIM * HID, HID, DIM, DIM);
    }
    EWL(k_wcvt, D_IP_PAD * DIM, in_proj_w, wt_ip, DIM, D_IN_PROJ, D_IP_PAD);
    EWL(k_wcvt, DIM * D_INNER, out_proj_w, wt_op, D_INNER, DIM, DIM);

    auto gemm = [&](const bf16* A, const bf16* Bt, float* Cc, int Nld, int K) {
        dim3 grid(Nld / 64, TOK / 256);
        k_gemm<<<grid, 256, 0, stream>>>(A, Bt, Cc, TOK, Nld, K);
    };

    auto mlp_block = [&](const bf16* w1T, const float* b1, const bf16* w2T,
                         const float* g, const float* b, const float* m, const float* v) {
        EWL(k_cvt, TOK * DIM, X, ab0, TOK * DIM);
        gemm(ab0, w1T, cf0, HID, DIM);
        EWL(k_bias_gelu, TOK * HID, cf0, b1, ab0, HID, TOK * HID);
        gemm(ab0, w2T, cf0, DIM, HID);
        EWL(k_bn_add, TOK * DIM, X, cf0, g, b, m, v, DIM, TOK * DIM);
    };

    // ---- forward ----
    mlp_block(wt_mlp0_w1, mlp0_b1, wt_mlp0_w2, mlp0_g, mlp0_b, mlp0_m, mlp0_v);

    for (int i = 0; i < 4; ++i) {
        // lfp: x @ lfp_w -> knn maxpool -> BN -> residual
        EWL(k_cvt, TOK * DIM, X, ab0, TOK * DIM);
        gemm(ab0, wt_lfp + (size_t)i * DIM * DIM, cf0, DIM, DIM);
        EWL(k_knn_bn_add, TOK * DIM, X, cf0, knn,
            lfp_g + (size_t)i * DIM, lfp_b + (size_t)i * DIM,
            lfp_m + (size_t)i * DIM, lfp_v + (size_t)i * DIM, TOK * DIM);

        // mamba2
        k_rms_cvt<<<TOK / 8, 256, 0, stream>>>(X, mb_norm, ab0);
        gemm(ab0, wt_ip, cf0, D_IP_PAD, DIM);                      // zxbcdt
        EWL(k_conv_silu, TOK * CONV_DIM, cf0, conv_w, conv_b, xbc, TOK * CONV_DIM);
        EWL(k_dt, TOK * NHEADS, cf0, dt_bias, dtv, TOK * NHEADS);
        k_scan<<<BB * NHEADS, 256, 0, stream>>>(xbc, dtv, A_log, Dp, ybuf);
        k_gate_rms<<<TOK / 8, 256, 0, stream>>>(ybuf, cf0, ssm_norm, ab0);
        gemm(ab0, wt_op, cf0, DIM, D_INNER);                       // out_proj
        EWL(k_add, TOK * DIM, X, cf0, TOK * DIM);                  // += x_res

        if (i & 1) {
            int j = i >> 1;
            mlp_block(wt_mlps_w1 + (size_t)j * HID * DIM, mlps_b1 + (size_t)j * HID,
                      wt_mlps_w2 + (size_t)j * DIM * HID,
                      mlps_g + (size_t)j * DIM, mlps_b + (size_t)j * DIM,
                      mlps_m + (size_t)j * DIM, mlps_v + (size_t)j * DIM);
        }
    }
}
